// SetAbstraction_22531398435368
// MI455X (gfx1250) — compile-verified, matrix operand order checked by reading, not run on hardware
//
#include <hip/hip_runtime.h>
#include <hip/hip_bf16.h>

typedef __attribute__((ext_vector_type(16))) _Float16 v16h;
typedef __attribute__((ext_vector_type(8)))  _Float16 v8h;
typedef __attribute__((ext_vector_type(8)))  float    v8f;

constexpr int Bb = 16;      // batch
constexpr int Nn = 4096;    // points
constexpr int Cc = 64;      // feature channels
constexpr int Hh = 128;     // hidden
constexpr int Ss = 1024;    // npoint
constexpr int Kk = 32;      // nsample
constexpr float R2 = 0.2f * 0.2f;

// ---------------------------------------------------------------------------
// Kernel 1: farthest point sampling. One block per batch. dist_min kept in LDS.
// Writes new_xyz (B,S,3) directly into d_out's first region.
// ---------------------------------------------------------------------------
__global__ __launch_bounds__(256) void fps_kernel(const float* __restrict__ xyz,
                                                  float* __restrict__ new_xyz) {
    __shared__ float dist[Nn];          // 16 KB
    __shared__ float wmax[8];
    __shared__ int   wimax[8];
    __shared__ float fx[3];

    const int b = blockIdx.x;
    const int t = threadIdx.x;
    const float* xb = xyz + (size_t)b * Nn * 3;

    const float x0 = xb[0], y0 = xb[1], z0 = xb[2];
    if (t == 0) {
        new_xyz[((size_t)b * Ss) * 3 + 0] = x0;
        new_xyz[((size_t)b * Ss) * 3 + 1] = y0;
        new_xyz[((size_t)b * Ss) * 3 + 2] = z0;
    }
    for (int i = t; i < Nn; i += 256) {
        float dx = xb[i * 3 + 0] - x0;
        float dy = xb[i * 3 + 1] - y0;
        float dz = xb[i * 3 + 2] - z0;
        dist[i] = dx * dx + dy * dy + dz * dz;
    }
    __syncthreads();

    for (int p = 1; p < Ss; ++p) {
        // per-thread argmax (strict > keeps first index inside a thread's stripe)
        float bv = -1.0f; int bi = 0;
        for (int i = t; i < Nn; i += 256) {
            float v = dist[i];
            if (v > bv) { bv = v; bi = i; }
        }
        // wave argmax, tie -> lower index
        for (int off = 16; off; off >>= 1) {
            float v2 = __shfl_xor(bv, off, 32);
            int   i2 = __shfl_xor(bi, off, 32);
            if (v2 > bv || (v2 == bv && i2 < bi)) { bv = v2; bi = i2; }
        }
        if ((t & 31) == 0) { wmax[t >> 5] = bv; wimax[t >> 5] = bi; }
        __syncthreads();
        if (t == 0) {
            float v = wmax[0]; int ix = wimax[0];
            for (int w = 1; w < 8; ++w)
                if (wmax[w] > v || (wmax[w] == v && wimax[w] < ix)) { v = wmax[w]; ix = wimax[w]; }
            float px = xb[ix * 3 + 0], py = xb[ix * 3 + 1], pz = xb[ix * 3 + 2];
            fx[0] = px; fx[1] = py; fx[2] = pz;
            new_xyz[((size_t)b * Ss + p) * 3 + 0] = px;
            new_xyz[((size_t)b * Ss + p) * 3 + 1] = py;
            new_xyz[((size_t)b * Ss + p) * 3 + 2] = pz;
        }
        __syncthreads();
        const float px = fx[0], py = fx[1], pz = fx[2];
        for (int i = t; i < Nn; i += 256) {
            float dx = xb[i * 3 + 0] - px;
            float dy = xb[i * 3 + 1] - py;
            float dz = xb[i * 3 + 2] - pz;
            float d2 = dx * dx + dy * dy + dz * dz;
            if (d2 < dist[i]) dist[i] = d2;
        }
        __syncthreads();
    }
}

// ---------------------------------------------------------------------------
// Kernel 2: ball query. One wave32 per (b,s). Selects nsample nearest within
// radius; ties (incl. out-of-ball "inf") resolved by lowest index, matching
// the reference's stable argsort.
// ---------------------------------------------------------------------------
__global__ __launch_bounds__(32) void ballq_kernel(const float* __restrict__ xyz,
                                                   const float* __restrict__ new_xyz,
                                                   int* __restrict__ gidx) {
    __shared__ float dist[Nn];          // 16 KB
    const int g = blockIdx.x;
    const int b = g >> 10;              // S = 1024
    const int lane = threadIdx.x;
    const float* xb = xyz + (size_t)b * Nn * 3;
    const float qx = new_xyz[(size_t)g * 3 + 0];
    const float qy = new_xyz[(size_t)g * 3 + 1];
    const float qz = new_xyz[(size_t)g * 3 + 2];
    const float INF = __int_as_float(0x7f800000);

    for (int i = lane; i < Nn; i += 32) {
        float dx = xb[i * 3 + 0] - qx;
        float dy = xb[i * 3 + 1] - qy;
        float dz = xb[i * 3 + 2] - qz;
        float d2 = dx * dx + dy * dy + dz * dz;
        dist[i] = (d2 <= R2) ? d2 : INF;
    }
    __syncthreads();

    for (int r = 0; r < Kk; ++r) {
        float bv = INF; int bi = 0x7fffffff;
        for (int i = lane; i < Nn; i += 32) {
            float v = dist[i];
            // taken entries are marked -1 and excluded; INF ties break by index
            if (v >= 0.0f && (v < bv || (v == bv && i < bi))) { bv = v; bi = i; }
        }
        for (int off = 16; off; off >>= 1) {
            float v2 = __shfl_xor(bv, off, 32);
            int   i2 = __shfl_xor(bi, off, 32);
            if (v2 < bv || (v2 == bv && i2 < bi)) { bv = v2; bi = i2; }
        }
        if (lane == 0) {
            gidx[(size_t)g * Kk + r] = bi;
            dist[bi] = -1.0f;           // mark taken
        }
        __syncthreads();
    }
}

// ---------------------------------------------------------------------------
// Kernel 3: pack a (Kreal x 128) f32 weight matrix into f16 WMMA B-fragments.
// Fragment layout (mirrors documented 16-bit A layout, transposed for B):
//   lane n (and n+16) hold column n; lanes<16: K=0..7,16..23; lanes>=16: K=8..15,24..31.
// dst[tile(kc*8+nt)][lane][halves 0..15], K padded with zeros.
// ---------------------------------------------------------------------------
__global__ __launch_bounds__(256) void pack_w(const float* __restrict__ W,
                                              _Float16* __restrict__ dst,
                                              int Kreal, int kcN) {
    int tid = blockIdx.x * 256 + threadIdx.x;
    int total = kcN * 8 * 32 * 16;
    if (tid >= total) return;
    int h    = tid & 15;
    int lane = (tid >> 4) & 31;
    int tile = tid >> 9;
    int kc = tile >> 3;
    int nt = tile & 7;
    int n = nt * 16 + (lane & 15);
    int k;
    if (lane < 16) k = kc * 32 + ((h < 8) ? h : h + 8);
    else           k = kc * 32 + ((h < 8) ? h + 8 : h + 16);
    float v = (k < Kreal) ? W[(size_t)k * 128 + n] : 0.0f;
    dst[tid] = (_Float16)v;
}

// ---------------------------------------------------------------------------
// Main kernel helpers
// ---------------------------------------------------------------------------
__device__ inline v16h load_a_frag(const _Float16* buf, int rs, int mt, int kc, int lane) {
    int row = mt * 16 + (lane & 15);
    int k0  = kc * 32 + ((lane < 16) ? 0 : 8);
    const _Float16* p = buf + row * rs + k0;
    v8h lo = *(const v8h*)p;
    v8h hi = *(const v8h*)(p + 16);
    return __builtin_shufflevector(lo, hi, 0, 1, 2, 3, 4, 5, 6, 7,
                                          8, 9, 10, 11, 12, 13, 14, 15);
}

template <int KC, int RS>
__device__ inline void gemm_relu_store(const _Float16* __restrict__ aBuf,
                                       const _Float16* __restrict__ wPacked,
                                       const float* __restrict__ bias,
                                       _Float16* __restrict__ outBuf, int lane) {
    const int cb = lane & 15;
    const int rb = (lane < 16) ? 0 : 8;
    for (int mt = 0; mt < 2; ++mt) {
        for (int nt = 0; nt < 8; ++nt) {
            float bn = bias[nt * 16 + cb];
            v8f c = {bn, bn, bn, bn, bn, bn, bn, bn};
#pragma unroll
            for (int kc = 0; kc < KC; ++kc) {
                v16h a = load_a_frag(aBuf, RS, mt, kc, lane);
                v16h w = *(const v16h*)(wPacked + ((size_t)(kc * 8 + nt) * 32 + lane) * 16);
                c = __builtin_amdgcn_wmma_f32_16x16x32_f16(false, a, false, w,
                                                           (short)0, c, false, false);
            }
            const int row0 = mt * 16 + rb;
            const int col  = nt * 16 + cb;
#pragma unroll
            for (int j = 0; j < 8; ++j) {
                float v = c[j];
                v = (v > 0.0f) ? v : 0.0f;
                outBuf[(row0 + j) * 128 + col] = (_Float16)v;
            }
        }
    }
}

// ---------------------------------------------------------------------------
// Kernel 4: fused set-abstraction MLP chain. One wave32 per (b,s) group,
// 2 waves per block. All four GEMM stages via V_WMMA_F32_16X16X32_F16.
// ---------------------------------------------------------------------------
#define WAVES 2
__global__ __launch_bounds__(WAVES * 32) void sa_main(
    const float* __restrict__ xyz, const float* __restrict__ feats,
    const float* __restrict__ b1, const float* __restrict__ b2,
    const float* __restrict__ b3, const float* __restrict__ b4,
    const _Float16* __restrict__ w1p, const _Float16* __restrict__ w2p,
    const _Float16* __restrict__ w3p, const _Float16* __restrict__ w4p,
    const int* __restrict__ gidx, const float* __restrict__ new_xyz,
    float* __restrict__ fout) {

    __shared__ __align__(16) _Float16 sA[WAVES][32 * 160];  // f_in / w_in (padded K)
    __shared__ __align__(16) _Float16 sH[WAVES][32 * 128];  // h / g
    __shared__ __align__(16) _Float16 sF[WAVES][32 * 128];  // f'
    __shared__ float sXN[WAVES][32 * 3];
    __shared__ float sM[WAVES][128];

    const int wv   = threadIdx.x >> 5;
    const int lane = threadIdx.x & 31;
    const int g    = blockIdx.x * WAVES + wv;
    const int b    = g >> 10;                        // S = 1024

    // ---- gather: lane k builds row k of f_in = [xyz_norm | feats], pad to 96
    {
        const int idx = gidx[(size_t)g * Kk + lane];
        const float* p = xyz + ((size_t)b * Nn + idx) * 3;
        const float nx = new_xyz[(size_t)g * 3 + 0];
        const float ny = new_xyz[(size_t)g * 3 + 1];
        const float nz = new_xyz[(size_t)g * 3 + 2];
        float dx = p[0] - nx, dy = p[1] - ny, dz = p[2] - nz;
        sXN[wv][lane * 3 + 0] = dx;
        sXN[wv][lane * 3 + 1] = dy;
        sXN[wv][lane * 3 + 2] = dz;
        _Float16* arow = &sA[wv][lane * 160];
        arow[0] = (_Float16)dx;
        arow[1] = (_Float16)dy;
        arow[2] = (_Float16)dz;
        const float4* f4 = (const float4*)(feats + ((size_t)b * Nn + idx) * Cc);
#pragma unroll
        for (int q = 0; q < Cc / 4; ++q) {
            float4 v = f4[q];
            arow[3 + q * 4 + 0] = (_Float16)v.x;
            arow[3 + q * 4 + 1] = (_Float16)v.y;
            arow[3 + q * 4 + 2] = (_Float16)v.z;
            arow[3 + q * 4 + 3] = (_Float16)v.w;
        }
#pragma unroll
        for (int q = 3 + Cc; q < 96; ++q) arow[q] = (_Float16)0.0f;
    }

    // ---- stage 1: h = relu(f_in @ W1 + b1)        (K = 96, 3 chunks)
    gemm_relu_store<3, 160>(sA[wv], w1p, b1, sH[wv], lane);

    // ---- stage 2: f' = relu(h @ W2 + b2)          (K = 128, 4 chunks)
    gemm_relu_store<4, 128>(sH[wv], w2p, b2, sF[wv], lane);

    // ---- mean over K rows (each lane owns 4 columns)
    {
        const int c0 = lane * 4;
#pragma unroll
        for (int cc = 0; cc < 4; ++cc) {
            float s = 0.0f;
            for (int r = 0; r < 32; ++r) s += (float)sF[wv][r * 128 + c0 + cc];
            sM[wv][c0 + cc] = s * (1.0f / 32.0f);
        }
    }

    // ---- build w_in = [xyz_norm | f' - mean], pad to 160 (lane = row)
    {
        _Float16* arow = &sA[wv][lane * 160];
        arow[0] = (_Float16)sXN[wv][lane * 3 + 0];
        arow[1] = (_Float16)sXN[wv][lane * 3 + 1];
        arow[2] = (_Float16)sXN[wv][lane * 3 + 2];
#pragma unroll 16
        for (int c = 0; c < 128; ++c)
            arow[3 + c] = (_Float16)((float)sF[wv][lane * 128 + c] - sM[wv][c]);
#pragma unroll
        for (int c = 131; c < 160; ++c) arow[c] = (_Float16)0.0f;
    }

    // ---- stage 3: g = relu(w_in @ W3 + b3)        (K = 160, 5 chunks)
    gemm_relu_store<5, 160>(sA[wv], w3p, b3, sH[wv], lane);

    // ---- stage 4: alpha = sigmoid(g @ W4 + b4); f_out = sum_k alpha * f'
    {
        const int cb = lane & 15;
        const int rb = (lane < 16) ? 0 : 8;
        for (int nt = 0; nt < 8; ++nt) {
            float acc = 0.0f;
            for (int mt = 0; mt < 2; ++mt) {
                float bn = b4[nt * 16 + cb];
                v8f c = {bn, bn, bn, bn, bn, bn, bn, bn};
#pragma unroll
                for (int kc = 0; kc < 4; ++kc) {
                    v16h a = load_a_frag(sH[wv], 128, mt, kc, lane);
                    v16h w = *(const v16h*)(w4p + ((size_t)(kc * 8 + nt) * 32 + lane) * 16);
                    c = __builtin_amdgcn_wmma_f32_16x16x32_f16(false, a, false, w,
                                                               (short)0, c, false, false);
                }
                const int row0 = mt * 16 + rb;
                const int col  = nt * 16 + cb;
#pragma unroll
                for (int j = 0; j < 8; ++j) {
                    float al = 1.0f / (1.0f + __expf(-c[j]));
                    acc += al * (float)sF[wv][(row0 + j) * 128 + col];
                }
            }
            float tot = acc + __shfl_xor(acc, 16, 32);
            if (lane < 16) fout[(size_t)g * Hh + nt * 16 + lane] = tot;
        }
    }
}

// ---------------------------------------------------------------------------
// Launch
// ---------------------------------------------------------------------------
extern "C" void kernel_launch(void* const* d_in, const int* in_sizes, int n_in,
                              void* d_out, int out_size, void* d_ws, size_t ws_size,
                              hipStream_t stream) {
    (void)in_sizes; (void)n_in; (void)out_size; (void)ws_size;
    const float* xyz   = (const float*)d_in[0];
    const float* feats = (const float*)d_in[1];
    const float* W1 = (const float*)d_in[2]; const float* b1 = (const float*)d_in[3];
    const float* W2 = (const float*)d_in[4]; const float* b2 = (const float*)d_in[5];
    const float* W3 = (const float*)d_in[6]; const float* b3 = (const float*)d_in[7];
    const float* W4 = (const float*)d_in[8]; const float* b4 = (const float*)d_in[9];

    float* out     = (float*)d_out;
    float* new_xyz = out;                       // (B,S,3)
    float* fout    = out + (size_t)Bb * Ss * 3; // (B,S,H)

    char* ws = (char*)d_ws;
    int* gidx = (int*)ws;                                 // B*S*K ints = 2 MB
    size_t off = (size_t)Bb * Ss * Kk * sizeof(int);
    _Float16* w1p = (_Float16*)(ws + off); off += (size_t)3 * 4096 * sizeof(_Float16);
    _Float16* w2p = (_Float16*)(ws + off); off += (size_t)4 * 4096 * sizeof(_Float16);
    _Float16* w3p = (_Float16*)(ws + off); off += (size_t)5 * 4096 * sizeof(_Float16);
    _Float16* w4p = (_Float16*)(ws + off);

    fps_kernel<<<Bb, 256, 0, stream>>>(xyz, new_xyz);

    pack_w<<<(3 * 4096 + 255) / 256, 256, 0, stream>>>(W1, w1p, 3 + Cc, 3);
    pack_w<<<(4 * 4096 + 255) / 256, 256, 0, stream>>>(W2, w2p, Hh, 4);
    pack_w<<<(5 * 4096 + 255) / 256, 256, 0, stream>>>(W3, w3p, 3 + Hh, 5);
    pack_w<<<(4 * 4096 + 255) / 256, 256, 0, stream>>>(W4, w4p, Hh, 4);

    ballq_kernel<<<Bb * Ss, 32, 0, stream>>>(xyz, new_xyz, gidx);

    sa_main<<<(Bb * Ss) / WAVES, WAVES * 32, 0, stream>>>(
        xyz, feats, b1, b2, b3, b4, w1p, w2p, w3p, w4p, gidx, new_xyz, fout);
}
